// MiniCPMMoE_66322884985035
// MI455X (gfx1250) — matrix operand order repaired and checked by reading, not compiled, not run on hardware
//
#include <hip/hip_runtime.h>
#include <hip/hip_bf16.h>

#define NE     8
#define TOPK   2
#define HID    2304
#define INTER  5760
#define NT     4096
#define NPAIR  (NT * TOPK)   // 8192

typedef __attribute__((ext_vector_type(16))) __bf16 v16bf;
typedef __attribute__((ext_vector_type(8)))  __bf16 v8bf;
typedef __attribute__((ext_vector_type(8)))  float  v8f;
typedef __attribute__((ext_vector_type(4)))  float  v4f;

static __device__ __forceinline__ int imin(int a, int b) { return a < b ? a : b; }

// fast sigmoid: one v_exp_f32 + one v_rcp_f32 (result goes to bf16 anyway)
static __device__ __forceinline__ float fast_sigmoid(float x) {
    return __builtin_amdgcn_rcpf(1.0f + __expf(-x));
}

// ---------------------------------------------------------------- zero init
__global__ void k_zero(float* __restrict__ out, int* __restrict__ counts) {
    size_t i = (size_t)blockIdx.x * blockDim.x + threadIdx.x;
    if (i < (size_t)NT * HID) out[i] = 0.0f;
    if (i < NE) counts[i] = 0;
}

// ------------------------------------------------------- fp32 -> bf16 cast
__global__ void k_cvt(const float* __restrict__ hs, __bf16* __restrict__ hbf) {
    size_t i = (size_t)blockIdx.x * blockDim.x + threadIdx.x;
    if (i < (size_t)NT * HID) hbf[i] = (__bf16)hs[i];
}

// ------------------------------------------------------------------ router
__global__ void k_route(const float* __restrict__ hs, const float* __restrict__ gw,
                        int* __restrict__ topk_id, float* __restrict__ topk_w,
                        int* __restrict__ counts) {
    int wave = threadIdx.x >> 5;
    int lane = threadIdx.x & 31;
    int t = blockIdx.x * 8 + wave;
    if (t >= NT) return;

    const float* x = hs + (size_t)t * HID;
    float acc[NE];
#pragma unroll
    for (int e = 0; e < NE; ++e) acc[e] = 0.0f;

    for (int k = lane; k < HID; k += 32) {
        float xv = x[k];
#pragma unroll
        for (int e = 0; e < NE; ++e) acc[e] += xv * gw[e * HID + k];
    }
#pragma unroll
    for (int e = 0; e < NE; ++e) {
#pragma unroll
        for (int off = 16; off >= 1; off >>= 1)
            acc[e] += __shfl_xor(acc[e], off, 32);
    }
    if (lane == 0) {
        int i1 = 0; float l1 = acc[0];
#pragma unroll
        for (int e = 1; e < NE; ++e) if (acc[e] > l1) { l1 = acc[e]; i1 = e; }
        int i2 = -1; float l2 = -3.0e38f;
#pragma unroll
        for (int e = 0; e < NE; ++e) if (e != i1 && acc[e] > l2) { l2 = acc[e]; i2 = e; }
        // softmax denominators cancel in the top-2 renormalization:
        float p  = __expf(l2 - l1);            // <= 1
        float w1 = __builtin_amdgcn_rcpf(1.0f + p);
        float w2 = p * w1;
        topk_id[t * 2 + 0] = i1;  topk_id[t * 2 + 1] = i2;
        topk_w[t * 2 + 0] = w1;   topk_w[t * 2 + 1] = w2;
        atomicAdd(&counts[i1], 1);
        atomicAdd(&counts[i2], 1);
    }
}

// ----------------------------------------------------------- prefix scan(8)
__global__ void k_scan(const int* __restrict__ counts, int* __restrict__ offsets,
                       int* __restrict__ cursor) {
    if (threadIdx.x == 0) {
        int o = 0;
        for (int e = 0; e < NE; ++e) { offsets[e] = o; cursor[e] = o; o += counts[e]; }
    }
}

// --------------------------------------------------------- token scatter
__global__ void k_scatter(const int* __restrict__ topk_id, const float* __restrict__ topk_w,
                          int* __restrict__ cursor, int* __restrict__ perm,
                          float* __restrict__ pw) {
    int i = blockIdx.x * blockDim.x + threadIdx.x;
    if (i >= NPAIR) return;
    int e = topk_id[i];
    int pos = atomicAdd(&cursor[e], 1);
    perm[pos] = i >> 1;          // token id
    pw[pos]   = topk_w[i];
}

// -------------------------------------------------- WMMA operand builders
static __device__ __forceinline__ v16bf load_a_bf16(const __bf16* p) {
    // A 16x32 bf16 layout: per-lane 16 contiguous K values (matches B)
    v8bf a0 = *(const v8bf*)(p);
    v8bf a1 = *(const v8bf*)(p + 8);
    v16bf a;
#pragma unroll
    for (int j = 0; j < 8; ++j) { a[j] = a0[j]; a[8 + j] = a1[j]; }
    return a;
}

static __device__ __forceinline__ v16bf load_b_cvt(const float* p) {
    // B 32x16: per-lane 16 contiguous K values, fp32 -> bf16 in regs
    v16bf b;
#pragma unroll
    for (int j = 0; j < 16; j += 4) {
        v4f f = *(const v4f*)(p + j);
        b[j] = (__bf16)f[0]; b[j+1] = (__bf16)f[1]; b[j+2] = (__bf16)f[2]; b[j+3] = (__bf16)f[3];
    }
    return b;
}

// ----------------------------------------------------------------- GEMM1
// gate_up GEMM + SiLU*up fusion.  block = 256 threads = 8 waves.
// wave tile: 16 tokens x 16 inter-neurons, K over HID in steps of 32 (bf16 WMMA).
// grid: x = INTER/128 (45), y = token tiles (256), z = expert (8)
__global__ void k_gemm1(const float* __restrict__ wsw,   // [NE][2*INTER][HID] fp32
                        const __bf16* __restrict__ hbf,  // [NT][HID] bf16
                        const int* __restrict__ perm, const int* __restrict__ offsets,
                        const int* __restrict__ counts,
                        __bf16* __restrict__ hbuf) {     // [NPAIR][INTER] bf16
    int e   = blockIdx.z;
    int cnt = counts[e];
    int mt  = blockIdx.y;
    if (mt * 16 >= cnt) return;            // wave-uniform exit
    int seg  = offsets[e];
    int wave = threadIdx.x >> 5;
    int lane = threadIdx.x & 31;
    int n0   = blockIdx.x * 128 + wave * 16;   // neuron base in [0, INTER)
    int lr   = lane & 15;
    int lhi  = lane >> 4;

    int mrow  = imin(mt * 16 + lr, cnt - 1);
    int token = perm[seg + mrow];
    const __bf16* arow = hbf + (size_t)token * HID + lhi * 16;
    const float*  bg   = wsw + (size_t)e * (2 * (size_t)INTER * HID)
                             + (size_t)(n0 + lr) * HID + lhi * 16;
    const float*  bu   = bg + (size_t)INTER * HID;

    v8f accg = {}; v8f accu = {};
#pragma unroll 2
    for (int k0 = 0; k0 < HID; k0 += 32) {
        v16bf a = load_a_bf16(arow + k0);
        v16bf b = load_b_cvt(bg + k0);
        v16bf c = load_b_cvt(bu + k0);
        accg = __builtin_amdgcn_wmma_f32_16x16x32_bf16(false, a, false, b, (short)0, accg, false, false);
        accu = __builtin_amdgcn_wmma_f32_16x16x32_bf16(false, a, false, c, (short)0, accu, false, false);
    }

    // D layout: acc[v] -> row (v + lhi*8), col lr.  h = silu(g)*u -> bf16
    __bf16* orow = hbuf + (size_t)seg * INTER + n0 + lr;
    int rbase = mt * 16 + lhi * 8;
    if (mt * 16 + 16 <= cnt) {             // full tile: uniform, no guards
#pragma unroll
        for (int v = 0; v < 8; ++v) {
            float g = accg[v];
            float h = g * fast_sigmoid(g) * accu[v];
            orow[(size_t)(rbase + v) * INTER] = (__bf16)h;
        }
    } else {                               // ragged last tile of this expert
#pragma unroll
        for (int v = 0; v < 8; ++v) {
            if (rbase + v < cnt) {
                float g = accg[v];
                float h = g * fast_sigmoid(g) * accu[v];
                orow[(size_t)(rbase + v) * INTER] = (__bf16)h;
            }
        }
    }
}

// ----------------------------------------------------------------- GEMM2
// down proj + weighted scatter-add into out.
// grid: x = HID/128 (18), y = token tiles (256), z = expert (8)
__global__ void k_gemm2(const float* __restrict__ w2,    // [NE][HID][INTER] fp32
                        const __bf16* __restrict__ hbuf, // [NPAIR][INTER] bf16
                        const int* __restrict__ perm, const float* __restrict__ pw,
                        const int* __restrict__ offsets, const int* __restrict__ counts,
                        float* __restrict__ out) {       // [NT][HID] fp32
    int e   = blockIdx.z;
    int cnt = counts[e];
    int mt  = blockIdx.y;
    if (mt * 16 >= cnt) return;
    int seg  = offsets[e];
    int wave = threadIdx.x >> 5;
    int lane = threadIdx.x & 31;
    int n0   = blockIdx.x * 128 + wave * 16;   // hidden col base
    int lr   = lane & 15;
    int lhi  = lane >> 4;

    int mrow = imin(mt * 16 + lr, cnt - 1);
    const __bf16* arow = hbuf + (size_t)(seg + mrow) * INTER + lhi * 16;
    const float*  brow = w2 + (size_t)e * HID * INTER + (size_t)(n0 + lr) * INTER + lhi * 16;

    v8f acc = {};
#pragma unroll 2
    for (int k0 = 0; k0 < INTER; k0 += 32) {
        v16bf a = load_a_bf16(arow + k0);
        v16bf b = load_b_cvt(brow + k0);
        acc = __builtin_amdgcn_wmma_f32_16x16x32_bf16(false, a, false, b, (short)0, acc, false, false);
    }

    int rbase = mt * 16 + lhi * 8;
    if (mt * 16 + 16 <= cnt) {             // full tile: uniform path
#pragma unroll
        for (int v = 0; v < 8; ++v) {
            int   token = perm[seg + rbase + v];
            float w     = pw[seg + rbase + v];
            atomicAdd(&out[(size_t)token * HID + n0 + lr], w * acc[v]);
        }
    } else {
#pragma unroll
        for (int v = 0; v < 8; ++v) {
            if (rbase + v < cnt) {
                int   token = perm[seg + rbase + v];
                float w     = pw[seg + rbase + v];
                atomicAdd(&out[(size_t)token * HID + n0 + lr], w * acc[v]);
            }
        }
    }
}

// ----------------------------------------------------------------- launch
extern "C" void kernel_launch(void* const* d_in, const int* in_sizes, int n_in,
                              void* d_out, int out_size, void* d_ws, size_t ws_size,
                              hipStream_t stream) {
    const float* hs  = (const float*)d_in[0];   // [NT][HID]
    const float* gw  = (const float*)d_in[1];   // [NE][HID]
    const float* wsw = (const float*)d_in[2];   // [NE][2*INTER][HID]
    const float* w2  = (const float*)d_in[3];   // [NE][HID][INTER]
    float* out = (float*)d_out;

    // carve scratch (256B aligned)
    char* base = (char*)d_ws;
    size_t off = 0;
    auto take = [&](size_t bytes) -> char* {
        char* p = base + off;
        off = (off + bytes + 255) & ~(size_t)255;
        return p;
    };
    __bf16* hbf     = (__bf16*)take((size_t)NT * HID * 2);        // ~18.9 MB
    __bf16* hbuf    = (__bf16*)take((size_t)NPAIR * INTER * 2);   // ~94.4 MB
    int*    topk_id = (int*)   take((size_t)NPAIR * 4);
    float*  topk_w  = (float*) take((size_t)NPAIR * 4);
    int*    counts  = (int*)   take(NE * 4);
    int*    offsets = (int*)   take(NE * 4);
    int*    cursor  = (int*)   take(NE * 4);
    int*    perm    = (int*)   take((size_t)NPAIR * 4);
    float*  pwv     = (float*) take((size_t)NPAIR * 4);
    (void)ws_size; (void)in_sizes; (void)n_in; (void)out_size;

    const int nelem = NT * HID;
    k_zero<<<(nelem + 255) / 256, 256, 0, stream>>>(out, counts);
    k_cvt <<<(nelem + 255) / 256, 256, 0, stream>>>(hs, hbf);
    k_route<<<NT / 8, 256, 0, stream>>>(hs, gw, topk_id, topk_w, counts);
    k_scan<<<1, 32, 0, stream>>>(counts, offsets, cursor);
    k_scatter<<<(NPAIR + 255) / 256, 256, 0, stream>>>(topk_id, topk_w, cursor, perm, pwv);

    dim3 g1(INTER / 128, NPAIR / 2 / 16, NE);  // (45, 256, 8)
    k_gemm1<<<g1, 256, 0, stream>>>(wsw, hbf, perm, offsets, counts, hbuf);

    dim3 g2(HID / 128, NPAIR / 2 / 16, NE);    // (18, 256, 8)
    k_gemm2<<<g2, 256, 0, stream>>>(w2, hbuf, perm, pwv, offsets, counts, out);
}
